// NeuralComplexityLoss_3178275799275
// MI455X (gfx1250) — compile-verified, matrix operand order checked by reading, not run on hardware
//
#include <hip/hip_runtime.h>
#include <hip/hip_bf16.h>
#include <math.h>

typedef __attribute__((ext_vector_type(16))) _Float16 v16h;
typedef __attribute__((ext_vector_type(8)))  float    v8f;

#define T_LEN 1024
#define N_TPL 1022          // T - M, M = 2
#define R_TH  0.2f
#define TILE  128
#define SUB   8             // each thread: 8x8 pair sub-tile (16x16 thread grid)

// Sum v across all 32 lanes of the wave using V_WMMA_F32_16X16X32_F16.
// A-layout (16-bit, 16x32): lanes 0-15 hold A[lane][K=0] in VGPR0 low half,
// lanes 16-31 hold A[lane-16][K=8].  With B == all-ones:
//   D[M][n] = v[M] + v[M+16]  for every column n.
// C/D layout: lane L<16, VGPR r -> D[r][L]; lane L>=16 -> D[r+8][L-16].
// So sum of the 8 D regs in lane 0 + lane 16 = full 32-lane sum (exact: small ints).
__device__ __forceinline__ float wave_wmma_sum(float v) {
  v16h a = {};
  a[0] = (_Float16)v;
  v16h b;
#pragma unroll
  for (int k = 0; k < 16; ++k) b[k] = (_Float16)1.0f;
  v8f c = {};
  c = __builtin_amdgcn_wmma_f32_16x16x32_f16(false, a, false, b,
                                             (short)0, c, false, false);
  float part = 0.0f;
#pragma unroll
  for (int r = 0; r < 8; ++r) part += c[r];
  return __shfl(part, 0, 32) + __shfl(part, 16, 32);
}

__global__ void __launch_bounds__(256) sampen_zero_kernel(unsigned int* cnt) {
  cnt[threadIdx.x] = 0u;   // 256 counters: 128 signals x {m, m1}
}

__global__ void __launch_bounds__(256) sampen_count_kernel(
    const float* __restrict__ pred, const float* __restrict__ targ,
    unsigned int* __restrict__ cnt) {
  __shared__ float xs[T_LEN];
  __shared__ float red[256];

  const int t   = threadIdx.x;
  const int sig = blockIdx.y;                       // 0..127
  const float* src = (sig < 64) ? (pred + sig * T_LEN)
                                : (targ + (sig - 64) * T_LEN);

  // ---- load raw signal (4 elements / thread) ----
  float xr[4];
#pragma unroll
  for (int k = 0; k < 4; ++k) xr[k] = src[t + 256 * k];

  // ---- mean ----
  red[t] = xr[0] + xr[1] + xr[2] + xr[3];
  __syncthreads();
  for (int off = 128; off > 0; off >>= 1) {
    if (t < off) red[t] += red[t + off];
    __syncthreads();
  }
  const float mean = red[0] * (1.0f / (float)T_LEN);
  __syncthreads();

  // ---- variance (ddof = 1) ----
  float q = 0.0f;
#pragma unroll
  for (int k = 0; k < 4; ++k) { float d = xr[k] - mean; q += d * d; }
  red[t] = q;
  __syncthreads();
  for (int off = 128; off > 0; off >>= 1) {
    if (t < off) red[t] += red[t + off];
    __syncthreads();
  }
  const float inv = 1.0f / (sqrtf(red[0] * (1.0f / (float)(T_LEN - 1))) + 1e-8f);
  __syncthreads();

  // ---- normalized signal into LDS ----
#pragma unroll
  for (int k = 0; k < 4; ++k) xs[t + 256 * k] = (xr[k] - mean) * inv;
  __syncthreads();

  // ---- 128x128 pair tile of the N x N template-distance matrix ----
  const int tile = blockIdx.x;                      // 0..63 (8x8 tiles)
  const int i0 = (tile >> 3) * TILE;
  const int j0 = (tile & 7) * TILE;
  const int ty = t >> 4, tx = t & 15;
  const int ib = i0 + ty * SUB;
  const int jb = j0 + tx * SUB;

  float xi[SUB + 2], xj[SUB + 2];
#pragma unroll
  for (int k = 0; k < SUB + 2; ++k) {
    xi[k] = xs[min(ib + k, T_LEN - 1)];
    xj[k] = xs[min(jb + k, T_LEN - 1)];
  }

  int cm = 0, cm1 = 0;
#pragma unroll
  for (int a = 0; a < SUB; ++a) {
    const bool iok = (ib + a) < N_TPL;
#pragma unroll
    for (int b = 0; b < SUB; ++b) {
      const bool ok = iok && ((jb + b) < N_TPL);
      const float d0 = fabsf(xi[a]     - xj[b]);
      const float d1 = fabsf(xi[a + 1] - xj[b + 1]);
      const float d2 = fabsf(xi[a + 2] - xj[b + 2]);
      const float dm  = fmaxf(d0, d1);
      const float dm1 = fmaxf(dm, d2);
      cm  += (ok && (dm  <= R_TH)) ? 1 : 0;   // predicated, no divergence
      cm1 += (ok && (dm1 <= R_TH)) ? 1 : 0;
    }
  }

  // ---- per-wave reduction via WMMA (EXEC is all-ones here) ----
  const float tm  = wave_wmma_sum((float)cm);   // per-lane counts <= 64: exact f16
  const float tm1 = wave_wmma_sum((float)cm1);
  if ((t & 31) == 0) {
    atomicAdd(&cnt[2 * sig + 0], (unsigned int)(tm  + 0.5f));
    atomicAdd(&cnt[2 * sig + 1], (unsigned int)(tm1 + 0.5f));
  }
}

__device__ __forceinline__ float sampen_ent(unsigned int m, unsigned int m1) {
  const float ratio = (float)m1 / (float)(m > 0u ? m : 1u);
  const float e = -logf(fmaxf(ratio, 1e-30f));
  return (m > 0u && m1 > 0u) ? e : 0.0f;
}

__global__ void __launch_bounds__(64) sampen_finalize_kernel(
    const unsigned int* __restrict__ cnt, float* __restrict__ out) {
  __shared__ float red[64];
  const int t = threadIdx.x;                        // 0..63 channel pairs
  const unsigned int mP  = cnt[2 * t + 0];
  const unsigned int m1P = cnt[2 * t + 1];
  const unsigned int mT  = cnt[2 * (64 + t) + 0];
  const unsigned int m1T = cnt[2 * (64 + t) + 1];
  const float d = sampen_ent(mP, m1P) - sampen_ent(mT, m1T);
  red[t] = d * d;
  __syncthreads();
  for (int off = 32; off > 0; off >>= 1) {
    if (t < off) red[t] += red[t + off];
    __syncthreads();
  }
  if (t == 0) out[0] = red[0] * (1.0f / 64.0f);
}

extern "C" void kernel_launch(void* const* d_in, const int* in_sizes, int n_in,
                              void* d_out, int out_size, void* d_ws, size_t ws_size,
                              hipStream_t stream) {
  const float* pred = (const float*)d_in[0];   // [4,16,1024] f32
  const float* targ = (const float*)d_in[1];   // [4,16,1024] f32
  unsigned int* cnt = (unsigned int*)d_ws;     // 256 u32 counters (1 KB)
  float* out = (float*)d_out;                  // scalar f32

  sampen_zero_kernel<<<1, 256, 0, stream>>>(cnt);
  sampen_count_kernel<<<dim3(64, 128), 256, 0, stream>>>(pred, targ, cnt);
  sampen_finalize_kernel<<<1, 64, 0, stream>>>(cnt, out);
}
